// GATConv_77206332112931
// MI455X (gfx1250) — matrix operand order repaired and checked by reading, not compiled, not run on hardware
//
#include <hip/hip_runtime.h>

#define N_HEADS 4
#define DHEAD   64
#define HD      256
#define INDIM   256
#define EDIM    32
#define NEG_SLOPE 0.2f

typedef __attribute__((ext_vector_type(16))) __bf16 v16bf;
typedef __attribute__((ext_vector_type(8)))  float  v8f;

__device__ __forceinline__ float4 ld4(const float* p) {
    return *reinterpret_cast<const float4*>(p);
}

__device__ __forceinline__ void pack16(v16bf& v, float4 a, float4 b, float4 c, float4 d) {
    v[0]=(__bf16)a.x;  v[1]=(__bf16)a.y;  v[2]=(__bf16)a.z;  v[3]=(__bf16)a.w;
    v[4]=(__bf16)b.x;  v[5]=(__bf16)b.y;  v[6]=(__bf16)b.z;  v[7]=(__bf16)b.w;
    v[8]=(__bf16)c.x;  v[9]=(__bf16)c.y;  v[10]=(__bf16)c.z; v[11]=(__bf16)c.w;
    v[12]=(__bf16)d.x; v[13]=(__bf16)d.y; v[14]=(__bf16)d.z; v[15]=(__bf16)d.w;
}

// ---------------------------------------------------------------------------
// Kernel 1: fold edge MLP + att_edge into W_eff[c][h] = sum_d linE[h*64+d][c]*attE[h][d]
// ---------------------------------------------------------------------------
__global__ void weff_kernel(const float* __restrict__ linE,   // [256,32]
                            const float* __restrict__ attE,   // [4,64]
                            float* __restrict__ weff) {       // [32,4]
    int t = threadIdx.x;                  // 0..127
    if (t >= EDIM * N_HEADS) return;
    int c = t & (EDIM - 1);
    int h = t >> 5;
    float s = 0.f;
    for (int d = 0; d < DHEAD; ++d)
        s += linE[(h * DHEAD + d) * EDIM + c] * attE[h * DHEAD + d];
    weff[c * N_HEADS + h] = s;
}

// ---------------------------------------------------------------------------
// Kernel 2: x_proj = x @ W^T via v_wmma_f32_16x16x32_bf16.
// One wave computes a 16x64 tile: 4 accumulators share each A-fragment,
// so x is streamed 4x total instead of 16x. 4 back-to-back WMMA per K-step.
// ---------------------------------------------------------------------------
__global__ __launch_bounds__(32) void xproj_wmma_kernel(
    const float* __restrict__ x, const float* __restrict__ W,
    float* __restrict__ xp, int nNodes) {
    const int tm   = blockIdx.x >> 2;   // row tile (16 nodes)
    const int tg   = blockIdx.x & 3;    // column group (64 channels)
    const int lane = threadIdx.x & 31;
    const int half = lane >> 4;         // 0 | 1
    const int l15  = lane & 15;

    int arowIdx = tm * 16 + l15;
    if (arowIdx >= nNodes) arowIdx = nNodes - 1;            // clamp loads, EXEC all-1
    const float* arow = x + (size_t)arowIdx * INDIM;
    // B[k][n] = W[n][k]; lane l15 owns column (tg*64 + j*16 + l15)
    const float* brow = W + (size_t)(tg * 64 + l15) * INDIM;

    v8f acc0 = {}, acc1 = {}, acc2 = {}, acc3 = {};
    for (int k0 = 0; k0 < INDIM; k0 += 32) {
        // A fragment (16x32 bf16): elems 0..15 -> K=ka..ka+15
        const int ka = k0 + half * 16;
        // B fragment (32x16 bf16): elems 0..15 -> K=kb..kb+15
        const int kb = k0 + half * 16;
        v16bf av;
        pack16(av, ld4(arow + ka),      ld4(arow + ka + 4),
                   ld4(arow + ka + 8),  ld4(arow + ka + 12));
        const float* bj = brow + kb;
        v16bf bv0, bv1, bv2, bv3;
        pack16(bv0, ld4(bj),            ld4(bj + 4),           ld4(bj + 8),           ld4(bj + 12));
        bj += 16 * INDIM;
        pack16(bv1, ld4(bj),            ld4(bj + 4),           ld4(bj + 8),           ld4(bj + 12));
        bj += 16 * INDIM;
        pack16(bv2, ld4(bj),            ld4(bj + 4),           ld4(bj + 8),           ld4(bj + 12));
        bj += 16 * INDIM;
        pack16(bv3, ld4(bj),            ld4(bj + 4),           ld4(bj + 8),           ld4(bj + 12));
        acc0 = __builtin_amdgcn_wmma_f32_16x16x32_bf16(false, av, false, bv0, (short)0, acc0, false, false);
        acc1 = __builtin_amdgcn_wmma_f32_16x16x32_bf16(false, av, false, bv1, (short)0, acc1, false, false);
        acc2 = __builtin_amdgcn_wmma_f32_16x16x32_bf16(false, av, false, bv2, (short)0, acc2, false, false);
        acc3 = __builtin_amdgcn_wmma_f32_16x16x32_bf16(false, av, false, bv3, (short)0, acc3, false, false);
    }

    // D layout: VGPR r, lanes 0-15 -> M=r, lanes 16-31 -> M=r+8; N = lane&15
    const int rowBase = tm * 16 + half * 8;
    float* obase = xp + (size_t)rowBase * HD + tg * 64 + l15;
    if (tm * 16 + 16 <= nNodes) {       // wave-uniform fast path: no guards
        #pragma unroll
        for (int r = 0; r < 8; ++r) {
            float* orow = obase + (size_t)r * HD;
            orow[0]  = acc0[r];
            orow[16] = acc1[r];
            orow[32] = acc2[r];
            orow[48] = acc3[r];
        }
    } else {
        #pragma unroll
        for (int r = 0; r < 8; ++r) {
            if (rowBase + r < nNodes) {
                float* orow = obase + (size_t)r * HD;
                orow[0]  = acc0[r];
                orow[16] = acc1[r];
                orow[32] = acc2[r];
                orow[48] = acc3[r];
            }
        }
    }
}

// ---------------------------------------------------------------------------
// Kernel 3: per-node attention scores  s_src[n][h], s_dst[n][h]
// ---------------------------------------------------------------------------
__global__ void score_kernel(const float* __restrict__ xp,
                             const float* __restrict__ attS,
                             const float* __restrict__ attD,
                             float* __restrict__ ssrc, float* __restrict__ sdst,
                             int total) {
    int idx = blockIdx.x * blockDim.x + threadIdx.x;
    if (idx >= total) return;
    int node = idx >> 2, h = idx & 3;
    const float* row = xp + (size_t)node * HD + h * DHEAD;
    const float* as  = attS + h * DHEAD;
    const float* ad  = attD + h * DHEAD;
    float s0 = 0.f, s1 = 0.f;
    for (int d = 0; d < DHEAD; d += 4) {
        float4 v = ld4(row + d), a = ld4(as + d), b = ld4(ad + d);
        s0 += v.x * a.x + v.y * a.y + v.z * a.z + v.w * a.w;
        s1 += v.x * b.x + v.y * b.y + v.z * b.z + v.w * b.w;
    }
    ssrc[idx] = s0; sdst[idx] = s1;
}

// ---------------------------------------------------------------------------
// Kernel 4a: out = bias (broadcast); 4b: seg-max keys / denom = 0
// ---------------------------------------------------------------------------
__global__ void init_out_kernel(float* __restrict__ out,
                                const float* __restrict__ bias) {
    out[(size_t)blockIdx.x * HD + threadIdx.x] = bias[threadIdx.x];
}

__global__ void init_stats_kernel(unsigned* __restrict__ segk,
                                  float* __restrict__ denom, int total) {
    int idx = blockIdx.x * blockDim.x + threadIdx.x;
    if (idx >= total) return;
    segk[idx]  = 0u;     // sentinel; every node has a self loop so it's always beaten
    denom[idx] = 0.f;
}

// monotone float -> uint key for atomicMax
__device__ __forceinline__ unsigned fkey(float f) {
    unsigned u = __float_as_uint(f);
    return (u & 0x80000000u) ? ~u : (u | 0x80000000u);
}

// ---------------------------------------------------------------------------
// Kernel 5: edge logits (+ folded edge term), leaky-relu, segment atomicMax
// ---------------------------------------------------------------------------
__global__ void att_logits_kernel(const int* __restrict__ ei,
                                  const float* __restrict__ ea,
                                  const float* __restrict__ weff,
                                  const float* __restrict__ ssrc,
                                  const float* __restrict__ sdst,
                                  float* __restrict__ attb,
                                  unsigned* __restrict__ segk,
                                  int E_, int total) {
    int idx = blockIdx.x * blockDim.x + threadIdx.x;
    if (idx >= total) return;
    int eid = idx >> 2, h = idx & 3;
    int src, dst; float a;
    if (eid < E_) {
        src = ei[eid]; dst = ei[E_ + eid];
        float t = 0.f;
        const float* erow = ea + (size_t)eid * EDIM;
        for (int c = 0; c < EDIM; ++c) t += erow[c] * weff[c * N_HEADS + h];
        a = ssrc[src * 4 + h] + sdst[dst * 4 + h] + t;
    } else {                                   // self loop, edge_attr == 0
        src = dst = eid - E_;
        a = ssrc[src * 4 + h] + sdst[dst * 4 + h];
    }
    a = (a >= 0.f) ? a : NEG_SLOPE * a;
    attb[idx] = a;
    atomicMax(&segk[dst * 4 + h], fkey(a));
}

// ---------------------------------------------------------------------------
// Kernel 6: decode seg-max keys to floats (in place)
// ---------------------------------------------------------------------------
__global__ void decode_max_kernel(unsigned* __restrict__ segk, int total) {
    int idx = blockIdx.x * blockDim.x + threadIdx.x;
    if (idx >= total) return;
    unsigned k = segk[idx];
    float m = (k & 0x80000000u) ? __uint_as_float(k ^ 0x80000000u)
                                : __uint_as_float(~k);
    reinterpret_cast<float*>(segk)[idx] = m;
}

// ---------------------------------------------------------------------------
// Kernel 7: exp(att - max) and denominator accumulation
// ---------------------------------------------------------------------------
__global__ void softmax_num_kernel(const int* __restrict__ ei,
                                   const float* __restrict__ segm,
                                   float* __restrict__ attb,
                                   float* __restrict__ denom,
                                   int E_, int total) {
    int idx = blockIdx.x * blockDim.x + threadIdx.x;
    if (idx >= total) return;
    int eid = idx >> 2, h = idx & 3;
    int dst = (eid < E_) ? ei[E_ + eid] : eid - E_;
    float ex = __expf(attb[idx] - segm[dst * 4 + h]);
    attb[idx] = ex;
    atomicAdd(&denom[dst * 4 + h], ex);
}

// ---------------------------------------------------------------------------
// Kernel 8: messages + scatter-sum; block per edge, thread per channel
// ---------------------------------------------------------------------------
__global__ __launch_bounds__(256) void scatter_kernel(
    const int* __restrict__ ei, const float* __restrict__ xp,
    const float* __restrict__ attb, const float* __restrict__ denom,
    float* __restrict__ out, int E_) {
    int eid = blockIdx.x;
    int c = threadIdx.x;            // 0..255
    int h = c >> 6;
    int src, dst;
    if (eid < E_) { src = ei[eid]; dst = ei[E_ + eid]; }
    else          { src = dst = eid - E_; }
    float alpha = attb[(size_t)eid * 4 + h] / (denom[dst * 4 + h] + 1e-12f);
    atomicAdd(&out[(size_t)dst * HD + c], xp[(size_t)src * HD + c] * alpha);
}

// ---------------------------------------------------------------------------
extern "C" void kernel_launch(void* const* d_in, const int* in_sizes, int n_in,
                              void* d_out, int out_size, void* d_ws, size_t ws_size,
                              hipStream_t stream) {
    const float* x    = (const float*)d_in[0];
    const int*   ei   = (const int*)  d_in[1];
    const float* ea   = (const float*)d_in[2];
    const float* linW = (const float*)d_in[3];
    const float* attS = (const float*)d_in[4];
    const float* attD = (const float*)d_in[5];
    const float* linE = (const float*)d_in[6];
    const float* attE = (const float*)d_in[7];
    const float* bias = (const float*)d_in[8];
    float* out = (float*)d_out;

    const int n  = in_sizes[0] / INDIM;   // 20000
    const int e  = in_sizes[1] / 2;       // 320000
    const int ep = e + n;                 // edges + self loops

    char* ws = (char*)d_ws;
    size_t off = 0;
    auto carve = [&](size_t bytes) -> void* {
        void* p = ws + off;
        off = (off + bytes + 255) & ~(size_t)255;
        return p;
    };
    float*    xp    = (float*)   carve((size_t)n * HD * sizeof(float));
    float*    ssrc  = (float*)   carve((size_t)n * N_HEADS * sizeof(float));
    float*    sdst  = (float*)   carve((size_t)n * N_HEADS * sizeof(float));
    float*    attb  = (float*)   carve((size_t)ep * N_HEADS * sizeof(float));
    unsigned* segk  = (unsigned*)carve((size_t)n * N_HEADS * sizeof(unsigned));
    float*    denom = (float*)   carve((size_t)n * N_HEADS * sizeof(float));
    float*    weff  = (float*)   carve(EDIM * N_HEADS * sizeof(float));
    (void)ws_size; (void)n_in; (void)out_size;

    const int nh  = n * N_HEADS;
    const int eph = ep * N_HEADS;
    const int rowTiles = (n + 15) / 16;

    weff_kernel<<<1, 128, 0, stream>>>(linE, attE, weff);
    xproj_wmma_kernel<<<rowTiles * 4, 32, 0, stream>>>(x, linW, xp, n);
    score_kernel<<<(nh + 255) / 256, 256, 0, stream>>>(xp, attS, attD, ssrc, sdst, nh);
    init_out_kernel<<<n, HD, 0, stream>>>(out, bias);
    init_stats_kernel<<<(nh + 255) / 256, 256, 0, stream>>>(segk, denom, nh);
    att_logits_kernel<<<(eph + 255) / 256, 256, 0, stream>>>(ei, ea, weff, ssrc, sdst,
                                                             attb, segk, e, eph);
    decode_max_kernel<<<(nh + 255) / 256, 256, 0, stream>>>(segk, nh);
    softmax_num_kernel<<<(eph + 255) / 256, 256, 0, stream>>>(ei, (const float*)segk,
                                                              attb, denom, e, eph);
    scatter_kernel<<<ep, 256, 0, stream>>>(ei, xp, attb, denom, out, e);
}
